// KGPathWalker_2525440770621
// MI455X (gfx1250) — matrix operand order repaired and checked by reading, not compiled, not run on hardware
//
#include <hip/hip_runtime.h>
#include <hip/hip_bf16.h>

// ---- CDNA5 WMMA types ----
typedef __attribute__((ext_vector_type(16))) __bf16 v16bf;
typedef __attribute__((ext_vector_type(8)))  float  v8f;
typedef __attribute__((ext_vector_type(4)))  unsigned int u32x4;

union Frag16 { v16bf v; u32x4 u[2]; };

#define DIMC 512
#define D2C  1024

__device__ __forceinline__ float sigmoidf_(float x) {
    return 1.0f / (1.0f + __expf(-x));
}

__device__ __forceinline__ v8f wmma_bf16(v16bf a, v16bf b, v8f c) {
    // D = A(16x32 bf16) x B(32x16 bf16) + C(16x16 f32)
    return __builtin_amdgcn_wmma_f32_16x16x32_bf16(
        false, a, false, b, (short)0, c, false, false);
}

// A-fragment (16x32 bf16) from LDS, row-major, row stride ldk (elements).
// Lanes 0-15: M=lane, K = kk+{0..7,16..23}; lanes 16-31: M=lane-16, K = kk+{8..15,24..31}.
__device__ __forceinline__ v16bf load_a_lds(const __bf16* As, int ldk, int kk, int lane) {
    int m  = lane & 15;
    int kb = (lane >> 4) << 3;            // 0 or 8
    const __bf16* p = As + m * ldk + kk + kb;
    Frag16 f;
    f.u[0] = *(const u32x4*)(p);          // K kb..kb+7
    f.u[1] = *(const u32x4*)(p + 16);     // K kb+16..kb+23
    return f.v;
}

// A-fragment direct from global fp32 rows, convert to bf16.
__device__ __forceinline__ v16bf load_a_glb(const float* __restrict__ A, int lda,
                                            int mrow0, int kk, int lane) {
    int m  = lane & 15;
    int kb = (lane >> 4) << 3;
    const float* p = A + (size_t)(mrow0 + m) * lda + kk + kb;
    v16bf r;
#pragma unroll
    for (int j = 0; j < 8; ++j) {
        r[j]     = (__bf16)p[j];
        r[8 + j] = (__bf16)p[16 + j];
    }
    return r;
}

// B-fragment (32x16 bf16) from pre-swizzled weights: fragment (kt,nt) is
// 32 lanes x 16 contiguous bf16 (two 16B global loads per lane, L2-resident).
__device__ __forceinline__ v16bf load_b_sw(const __bf16* __restrict__ Wsw,
                                           int ntiles, int kt, int nt, int lane) {
    const __bf16* p = Wsw + ((((kt * ntiles + nt) << 5) + lane) << 4);
    Frag16 f;
    f.u[0] = *(const u32x4*)(p);
    f.u[1] = *(const u32x4*)(p + 8);
    return f.v;
}

// ---- Pre-pass: swizzle fp32 weight [K,N] row-major into bf16 B-fragment layout ----
__global__ void swizzle_w(const float* __restrict__ W, __bf16* __restrict__ out,
                          int K, int N) {
    int t = blockIdx.x * blockDim.x + threadIdx.x;
    if (t >= K * N) return;
    int j    = t & 15;
    int lane = (t >> 4) & 31;
    int f    = t >> 9;
    int ntiles = N >> 4;
    int nt = f % ntiles;
    int kt = f / ntiles;
    int krow = (kt << 5) + ((lane >> 4) << 3) + (j & 7) + ((j >> 3) << 4);   // K index
    int ncol = (nt << 4) + (lane & 15);              // N index
    out[t] = (__bf16)W[(size_t)krow * N + ncol];
}

// ---- Fused kernel: 4 waves per block, one 16-row tile per block ----
// Each wave owns 8 N-tiles, processed as 2 groups of 4 independent accumulators.
__global__ __launch_bounds__(128)
void kg_fused(const float* __restrict__ x_bar,
              const float* __restrict__ h_t,
              const float* __restrict__ c_t,
              const __bf16* __restrict__ Wx_sw,  const __bf16* __restrict__ Wha_sw,
              const __bf16* __restrict__ Wxa_sw, const __bf16* __restrict__ Rel_sw,
              const __bf16* __restrict__ Wzc_sw, const __bf16* __restrict__ Whc_sw,
              const float* __restrict__ Wx_b,  const float* __restrict__ Wha_b,
              const float* __restrict__ Wxa_b, const float* __restrict__ Wzc_b,
              const float* __restrict__ Whc_b,
              const float* __restrict__ Whi_w, const float* __restrict__ Whi_b,
              const float* __restrict__ Wci_w, const float* __restrict__ Wci_b,
              const float* __restrict__ Wzo_w, const float* __restrict__ Wzo_b,
              const float* __restrict__ Who_w, const float* __restrict__ Who_b,
              const float* __restrict__ Wco_w, const float* __restrict__ Wco_b,
              float* __restrict__ out_h, float* __restrict__ out_c,
              float* __restrict__ out_r) {
    __shared__ __align__(16) __bf16 A_h[16 * DIMC];      // h tile, bf16
    __shared__ __align__(16) __bf16 A_x[16 * DIMC];      // x = x_bar@Wx
    __shared__ __align__(16) __bf16 A_al[16 * DIMC];     // alpha
    __shared__ __align__(16) __bf16 A_z[16 * DIMC];      // z = h + r
    __shared__ __align__(16) float  C_new[16 * DIMC];    // c_new, fp32
    __shared__ float i_lds[16];
    __shared__ float o_lds[16];

    const int tid   = threadIdx.x;
    const int lane  = tid & 31;
    const int w     = tid >> 5;             // wave id 0..3
    const int mbase = blockIdx.x << 4;
    const int n_l   = lane & 15;            // column within 16-wide tile (C/D layout)
    const int m_0   = (lane >> 4) << 3;     // lanes 16-31 hold M rows 8..15

    const v8f zero8 = {0.f, 0.f, 0.f, 0.f, 0.f, 0.f, 0.f, 0.f};

    // Stage h tile into LDS as bf16 (coalesced, all 128 threads)
    for (int idx = tid; idx < 16 * DIMC; idx += 128) {
        int row = idx >> 9, col = idx & (DIMC - 1);
        A_h[idx] = (__bf16)h_t[(size_t)(mbase + row) * DIMC + col];
    }
    __syncthreads();

    // ---- Phase 1: x = x_bar @ Wx + b  (K = 1024), A streamed from global ----
    for (int q = 0; q < 2; ++q) {
        const int ntb = (w << 3) + (q << 2);
        v8f acc[4] = {zero8, zero8, zero8, zero8};
        for (int kt = 0; kt < 32; ++kt) {
            v16bf a = load_a_glb(x_bar, D2C, mbase, kt << 5, lane);
#pragma unroll
            for (int j = 0; j < 4; ++j)
                acc[j] = wmma_bf16(a, load_b_sw(Wx_sw, 32, kt, ntb + j, lane), acc[j]);
        }
#pragma unroll
        for (int j = 0; j < 4; ++j) {
            int nt = ntb + j;
            float bias = Wx_b[(nt << 4) + n_l];
#pragma unroll
            for (int r = 0; r < 8; ++r)
                A_x[(m_0 + r) * DIMC + (nt << 4) + n_l] = (__bf16)(acc[j][r] + bias);
        }
    }
    __syncthreads();

    // ---- Phase 2: alpha = sigmoid(h@Wha + x@Wxa + b) ----
    for (int q = 0; q < 2; ++q) {
        const int ntb = (w << 3) + (q << 2);
        v8f acc[4] = {zero8, zero8, zero8, zero8};
        for (int kt = 0; kt < 16; ++kt) {
            v16bf a = load_a_lds(A_h, DIMC, kt << 5, lane);
#pragma unroll
            for (int j = 0; j < 4; ++j)
                acc[j] = wmma_bf16(a, load_b_sw(Wha_sw, 32, kt, ntb + j, lane), acc[j]);
        }
        for (int kt = 0; kt < 16; ++kt) {
            v16bf a = load_a_lds(A_x, DIMC, kt << 5, lane);
#pragma unroll
            for (int j = 0; j < 4; ++j)
                acc[j] = wmma_bf16(a, load_b_sw(Wxa_sw, 32, kt, ntb + j, lane), acc[j]);
        }
#pragma unroll
        for (int j = 0; j < 4; ++j) {
            int nt = ntb + j;
            float bias = Wha_b[(nt << 4) + n_l] + Wxa_b[(nt << 4) + n_l];
#pragma unroll
            for (int r = 0; r < 8; ++r)
                A_al[(m_0 + r) * DIMC + (nt << 4) + n_l] =
                    (__bf16)sigmoidf_(acc[j][r] + bias);
        }
    }
    __syncthreads();

    // ---- Phase 3: r = alpha @ relations ; z = h + r ----
    for (int q = 0; q < 2; ++q) {
        const int ntb = (w << 3) + (q << 2);
        v8f acc[4] = {zero8, zero8, zero8, zero8};
        for (int kt = 0; kt < 16; ++kt) {
            v16bf a = load_a_lds(A_al, DIMC, kt << 5, lane);
#pragma unroll
            for (int j = 0; j < 4; ++j)
                acc[j] = wmma_bf16(a, load_b_sw(Rel_sw, 32, kt, ntb + j, lane), acc[j]);
        }
#pragma unroll
        for (int j = 0; j < 4; ++j) {
            int nt = ntb + j;
#pragma unroll
            for (int r = 0; r < 8; ++r) {
                int row = m_0 + r, col = (nt << 4) + n_l;
                size_t g = (size_t)(mbase + row) * DIMC + col;
                float rv = acc[j][r];
                out_r[g] = rv;
                A_z[row * DIMC + col] = (__bf16)(h_t[g] + rv);
            }
        }
    }
    __syncthreads();

    // ---- Phase 4: i = sigmoid(h.Whi + c.Wci + b)  (8 lanes per row) ----
    {
        int row = (w << 2) + (lane & 3);    // 4 rows per wave
        int seg = lane >> 2;                // 8 segments of 64
        int k0  = seg << 6;
        size_t g0 = (size_t)(mbase + row) * DIMC + k0;
        float s = 0.f;
        for (int k = 0; k < 64; ++k)
            s += h_t[g0 + k] * Whi_w[k0 + k] + c_t[g0 + k] * Wci_w[k0 + k];
        s += __shfl_xor(s, 4, 32);
        s += __shfl_xor(s, 8, 32);
        s += __shfl_xor(s, 16, 32);
        if ((lane >> 2) == 0) i_lds[row] = sigmoidf_(s + Whi_b[0] + Wci_b[0]);
    }
    __syncthreads();

    // ---- Phase 5: c_new = (1-i)*c + i*tanh(z@Wzc + h@Whc + b) ----
    for (int q = 0; q < 2; ++q) {
        const int ntb = (w << 3) + (q << 2);
        v8f acc[4] = {zero8, zero8, zero8, zero8};
        for (int kt = 0; kt < 16; ++kt) {
            v16bf a = load_a_lds(A_z, DIMC, kt << 5, lane);
#pragma unroll
            for (int j = 0; j < 4; ++j)
                acc[j] = wmma_bf16(a, load_b_sw(Wzc_sw, 32, kt, ntb + j, lane), acc[j]);
        }
        for (int kt = 0; kt < 16; ++kt) {
            v16bf a = load_a_lds(A_h, DIMC, kt << 5, lane);
#pragma unroll
            for (int j = 0; j < 4; ++j)
                acc[j] = wmma_bf16(a, load_b_sw(Whc_sw, 32, kt, ntb + j, lane), acc[j]);
        }
#pragma unroll
        for (int j = 0; j < 4; ++j) {
            int nt = ntb + j;
            float bias = Wzc_b[(nt << 4) + n_l] + Whc_b[(nt << 4) + n_l];
#pragma unroll
            for (int r = 0; r < 8; ++r) {
                int row = m_0 + r, col = (nt << 4) + n_l;
                size_t g = (size_t)(mbase + row) * DIMC + col;
                float it = i_lds[row];
                float cn = (1.0f - it) * c_t[g] + it * tanhf(acc[j][r] + bias);
                out_c[g] = cn;
                C_new[row * DIMC + col] = cn;
            }
        }
    }
    __syncthreads();

    // ---- Phase 6: o = sigmoid(z.Wzo + h.Who + c_new.Wco + b); h_new = o*tanh(c_new) ----
    {
        int row = (w << 2) + (lane & 3);
        int seg = lane >> 2;
        int k0  = seg << 6;
        float s = 0.f;
        for (int k = 0; k < 64; ++k) {
            int kk = k0 + k;
            s += (float)A_z[row * DIMC + kk] * Wzo_w[kk]
               + (float)A_h[row * DIMC + kk] * Who_w[kk]
               + C_new[row * DIMC + kk] * Wco_w[kk];
        }
        s += __shfl_xor(s, 4, 32);
        s += __shfl_xor(s, 8, 32);
        s += __shfl_xor(s, 16, 32);
        if ((lane >> 2) == 0) o_lds[row] = sigmoidf_(s + Wzo_b[0] + Who_b[0] + Wco_b[0]);
    }
    __syncthreads();

    for (int idx = tid; idx < 16 * DIMC; idx += 128) {
        int row = idx >> 9;
        out_h[(size_t)(mbase + row) * DIMC + (idx & (DIMC - 1))] =
            o_lds[row] * tanhf(C_new[idx]);
    }
}

extern "C" void kernel_launch(void* const* d_in, const int* in_sizes, int n_in,
                              void* d_out, int out_size, void* d_ws, size_t ws_size,
                              hipStream_t stream) {
    const float* x_bar     = (const float*)d_in[0];
    const float* relations = (const float*)d_in[1];
    const float* h_t       = (const float*)d_in[2];
    const float* c_t       = (const float*)d_in[3];
    const float* Wx_w  = (const float*)d_in[4];  const float* Wx_b  = (const float*)d_in[5];
    const float* Whi_w = (const float*)d_in[6];  const float* Whi_b = (const float*)d_in[7];
    const float* Wci_w = (const float*)d_in[8];  const float* Wci_b = (const float*)d_in[9];
    const float* Wzc_w = (const float*)d_in[10]; const float* Wzc_b = (const float*)d_in[11];
    const float* Whc_w = (const float*)d_in[12]; const float* Whc_b = (const float*)d_in[13];
    const float* Wzo_w = (const float*)d_in[14]; const float* Wzo_b = (const float*)d_in[15];
    const float* Who_w = (const float*)d_in[16]; const float* Who_b = (const float*)d_in[17];
    const float* Wco_w = (const float*)d_in[18]; const float* Wco_b = (const float*)d_in[19];
    const float* Wha_w = (const float*)d_in[20]; const float* Wha_b = (const float*)d_in[21];
    const float* Wxa_w = (const float*)d_in[22]; const float* Wxa_b = (const float*)d_in[23];

    const int B = in_sizes[2] / DIMC;   // 32768

    // Workspace: swizzled bf16 weights (~3.7 MB)
    __bf16* Wx_sw  = (__bf16*)d_ws;
    __bf16* Wha_sw = Wx_sw  + (size_t)D2C * DIMC;
    __bf16* Wxa_sw = Wha_sw + (size_t)DIMC * DIMC;
    __bf16* Rel_sw = Wxa_sw + (size_t)DIMC * DIMC;
    __bf16* Wzc_sw = Rel_sw + (size_t)DIMC * DIMC;
    __bf16* Whc_sw = Wzc_sw + (size_t)DIMC * DIMC;

    const int thr = 256;
    swizzle_w<<<(D2C * DIMC + thr - 1) / thr, thr, 0, stream>>>(Wx_w, Wx_sw, D2C, DIMC);
    swizzle_w<<<(DIMC * DIMC + thr - 1) / thr, thr, 0, stream>>>(Wha_w, Wha_sw, DIMC, DIMC);
    swizzle_w<<<(DIMC * DIMC + thr - 1) / thr, thr, 0, stream>>>(Wxa_w, Wxa_sw, DIMC, DIMC);
    swizzle_w<<<(DIMC * DIMC + thr - 1) / thr, thr, 0, stream>>>(relations, Rel_sw, DIMC, DIMC);
    swizzle_w<<<(DIMC * DIMC + thr - 1) / thr, thr, 0, stream>>>(Wzc_w, Wzc_sw, DIMC, DIMC);
    swizzle_w<<<(DIMC * DIMC + thr - 1) / thr, thr, 0, stream>>>(Whc_w, Whc_sw, DIMC, DIMC);

    float* out_h = (float*)d_out;
    float* out_c = out_h + (size_t)B * DIMC;
    float* out_r = out_c + (size_t)B * DIMC;

    kg_fused<<<B / 16, 128, 0, stream>>>(
        x_bar, h_t, c_t,
        Wx_sw, Wha_sw, Wxa_sw, Rel_sw, Wzc_sw, Whc_sw,
        Wx_b, Wha_b, Wxa_b, Wzc_b, Whc_b,
        Whi_w, Whi_b, Wci_w, Wci_b,
        Wzo_w, Wzo_b, Who_w, Who_b, Wco_w, Wco_b,
        out_h, out_c, out_r);
}
